// CliffordJEPAModel_25563645346659
// MI455X (gfx1250) — matrix operand order repaired and checked by reading, not compiled, not run on hardware
//
#include <hip/hip_runtime.h>

// ---------------- types ----------------
typedef __bf16  bf16;
typedef float   v8f   __attribute__((ext_vector_type(8)));
typedef bf16    v16bf __attribute__((ext_vector_type(16)));
typedef bf16    v8bf  __attribute__((ext_vector_type(8)));

#define D_EMB   768
#define NB      8
#define BATCH   64
#define SEQ     256
#define ROWS    (BATCH * SEQ)      // 16384
#define N3D     (3 * D_EMB)        // 2304
#define KT_CNT  (D_EMB / 32)       // 24
#define KT_HALF (KT_CNT / 2)       // 12
#define NT_CNT  (N3D / 16)         // 144
#define DT_CNT  (D_EMB / 16)       // 48

// ---------------- device helpers ----------------
__device__ __forceinline__ float sigmoid_f(float x) {
    return 1.0f / (1.0f + __expf(-x));
}
__device__ __forceinline__ float tanh_f(float x) {
    float t = __expf(-2.0f * x);
    return (1.0f - t) / (1.0f + t);
}

// LDS byte offset of a __shared__ object (generic addr low 32 bits = LDS offset)
__device__ __forceinline__ unsigned lds_off(const void* p) {
    return (unsigned)(unsigned long long)(uintptr_t)p;
}

// CDNA5 async DMA: LDS[lds_byte] = MEM[gaddr .. +15], tracked by ASYNCcnt
__device__ __forceinline__ void async_b128_to_lds(unsigned lds_byte, const void* gaddr) {
    asm volatile("global_load_async_to_lds_b128 %0, %1, off"
                 :: "v"(lds_byte), "v"(gaddr) : "memory");
}
__device__ __forceinline__ void wait_async0() {
    asm volatile("s_wait_asynccnt 0x0" ::: "memory");
}

// A-fragment (16x32 bf16, M x K) per ISA layout:
// lane L: M = L&15 ; half = L>>4 ; elems 0..7  = K  kk+half*8    .. +7
//                                  elems 8..15 = K  kk+16+half*8 .. +7
__device__ __forceinline__ v16bf load_a_frag(const bf16* __restrict__ A, int lda,
                                             int mbase, int kk, int lane) {
    int m    = mbase + (lane & 15);
    int half = (lane >> 4) & 1;
    const bf16* r = A + (size_t)m * lda + kk + half * 8;
    union { v16bf v; v8bf h[2]; } u;
    u.h[0] = *(const v8bf*)(r);
    u.h[1] = *(const v8bf*)(r + 16);
    return u.v;
}

// ---------------- kernels ----------------

// Pack W [3D x D] (row-major, fp32) into WMMA-B fragment order (bf16):
// P[((kt*NT + nt)*32 + lane)*16 + i] = W[nt*16 + (lane&15)][kt*32 + (i&8)*2 + (lane<16?0:8) + (i&7)]
__global__ void pack_w_kernel(const float* __restrict__ W, bf16* __restrict__ P) {
    int idx = blockIdx.x * blockDim.x + threadIdx.x;
    const int total = KT_CNT * NT_CNT * 32 * 16;
    if (idx >= total) return;
    int i  = idx & 15;
    int L  = (idx >> 4) & 31;
    int t  = idx >> 9;             // kt*NT + nt
    int nt = t % NT_CNT;
    int kt = t / NT_CNT;
    int n  = nt * 16 + (L & 15);
    int k  = kt * 32 + (i & 8) * 2 + ((L < 16) ? 0 : 8) + (i & 7);
    P[idx] = (bf16)W[(size_t)n * D_EMB + k];
}

// X[row][d] = (bf16) emb[tok[row]][d]   (row = b*SEQ + s)
__global__ void gather_kernel(const int* __restrict__ tok,
                              const float* __restrict__ emb,
                              bf16* __restrict__ X) {
    int idx = blockIdx.x * blockDim.x + threadIdx.x;
    if (idx >= ROWS * D_EMB) return;
    int d   = idx % D_EMB;
    int row = idx / D_EMB;
    X[idx]  = (bf16)emb[(size_t)tok[row] * D_EMB + d];
}

// Gi = X @ Wih^T + bih  -> bf16 [ROWS x 3D].
// B fragments for the block's nt are staged once into LDS via async DMA;
// all 4 waves then feed WMMA from ds_load.
__global__ void gi_gemm_kernel(const bf16* __restrict__ X,
                               const bf16* __restrict__ P,
                               const float* __restrict__ bih,
                               bf16* __restrict__ Gi) {
    __shared__ __align__(16) bf16 Bs[KT_CNT * 512];   // 24 KB
    int lane = threadIdx.x & 31;
    int wave = threadIdx.x >> 5;
    int tid  = threadIdx.x;
    int nt   = blockIdx.y;                 // 0..143

    // stage all B tiles for this nt: 1536 x 16B chunks, 12 per thread
    unsigned bs_base = lds_off(&Bs[0]);
    #pragma unroll
    for (int c = 0; c < 12; ++c) {
        int chunk = tid + c * 128;
        int kt    = chunk >> 6;            // 64 chunks per kt-tile
        int rem   = chunk & 63;
        const bf16* g = P + ((size_t)kt * NT_CNT + nt) * 512 + rem * 8;
        async_b128_to_lds(bs_base + chunk * 16, g);
    }
    wait_async0();
    __syncthreads();

    int mt    = blockIdx.x * 4 + wave;     // 0..1023
    int mbase = mt * 16;
    const bf16* arow = X + (size_t)(mbase + (lane & 15)) * D_EMB;
    v8f acc = {};
    for (int kt = 0; kt < KT_CNT; ++kt) {
        v16bf a = load_a_frag(X, D_EMB, mbase, kt * 32, lane);
        v16bf b = *(const v16bf*)(&Bs[(size_t)kt * 512 + lane * 16]);
        if (kt + 1 < KT_CNT)
            __builtin_prefetch(arow + (kt + 1) * 32, 0, 3);
        acc = __builtin_amdgcn_wmma_f32_16x16x32_bf16(false, a, false, b,
                                                      (short)0, acc, false, false);
    }
    int n      = nt * 16 + (lane & 15);
    int half   = (lane >> 4) & 1;
    float bias = bih[n];
    #pragma unroll
    for (int j = 0; j < 8; ++j) {
        int m = mbase + j + half * 8;              // C layout: VGPR j -> M=j / j+8
        Gi[(size_t)m * N3D + n] = (bf16)(acc[j] + bias);
    }
}

__global__ void zero_h_kernel(float* __restrict__ hf, bf16* __restrict__ hb) {
    int idx = blockIdx.x * blockDim.x + threadIdx.x;
    if (idx >= BATCH * D_EMB) return;
    hf[idx] = 0.0f;
    hb[idx] = (bf16)0.0f;
}

// One GRU time step: Gh = h@Whh^T (3 WMMA tiles/wave) fused with gate math.
// Whh fragments (r/z/n for this dtile) staged via async DMA in two 36 KB halves.
__global__ void gru_step_kernel(const bf16* __restrict__ Gi, int s,
                                const float* __restrict__ hprev_f,
                                const bf16* __restrict__ hprev_b,
                                const bf16* __restrict__ Phh,
                                const float* __restrict__ bhh,
                                float* __restrict__ hnext_f,
                                bf16* __restrict__ hnext_b) {
    __shared__ __align__(16) bf16 Bs[KT_HALF * 3 * 512];   // 36 KB
    int lane  = threadIdx.x & 31;
    int wave  = threadIdx.x >> 5;   // batch-row tile 0..3
    int tid   = threadIdx.x;
    int dtile = blockIdx.x;         // 0..47
    int mbase = wave * 16;
    unsigned bs_base = lds_off(&Bs[0]);

    v8f ar = {}, az = {}, an = {};
    for (int hk = 0; hk < 2; ++hk) {
        int ktbase = hk * KT_HALF;
        // stage 12 kt x 3 gates: 2304 x 16B chunks, 18 per thread
        #pragma unroll
        for (int c = 0; c < 18; ++c) {
            int chunk = tid + c * 128;
            int blk   = chunk >> 6;         // (lkt*3 + g), 0..35
            int rem   = chunk & 63;
            int lkt   = blk / 3;
            int g     = blk % 3;
            const bf16* gp = Phh +
                ((size_t)(ktbase + lkt) * NT_CNT + g * DT_CNT + dtile) * 512 + rem * 8;
            async_b128_to_lds(bs_base + chunk * 16, gp);
        }
        wait_async0();
        __syncthreads();

        for (int lkt = 0; lkt < KT_HALF; ++lkt) {
            v16bf a  = load_a_frag(hprev_b, D_EMB, mbase, (ktbase + lkt) * 32, lane);
            v16bf br = *(const v16bf*)(&Bs[((size_t)lkt * 3 + 0) * 512 + lane * 16]);
            v16bf bz = *(const v16bf*)(&Bs[((size_t)lkt * 3 + 1) * 512 + lane * 16]);
            v16bf bn = *(const v16bf*)(&Bs[((size_t)lkt * 3 + 2) * 512 + lane * 16]);
            ar = __builtin_amdgcn_wmma_f32_16x16x32_bf16(false, a, false, br, (short)0, ar, false, false);
            az = __builtin_amdgcn_wmma_f32_16x16x32_bf16(false, a, false, bz, (short)0, az, false, false);
            an = __builtin_amdgcn_wmma_f32_16x16x32_bf16(false, a, false, bn, (short)0, an, false, false);
        }
        __syncthreads();   // Bs reused by next half
    }

    int d    = dtile * 16 + (lane & 15);
    int half = (lane >> 4) & 1;
    float b_r = bhh[d], b_z = bhh[D_EMB + d], b_n = bhh[2 * D_EMB + d];
    #pragma unroll
    for (int j = 0; j < 8; ++j) {
        int m = mbase + j + half * 8;                  // batch index 0..63
        size_t gir = ((size_t)m * SEQ + s) * N3D;
        float ir  = (float)Gi[gir + d];
        float iz  = (float)Gi[gir + D_EMB + d];
        float inn = (float)Gi[gir + 2 * D_EMB + d];
        float r  = sigmoid_f(ir + ar[j] + b_r);
        float z  = sigmoid_f(iz + az[j] + b_z);
        float nn = tanh_f(inn + r * (an[j] + b_n));
        float hp = hprev_f[(size_t)m * D_EMB + d];
        float hn = (1.0f - z) * nn + z * hp;
        hnext_f[(size_t)m * D_EMB + d] = hn;
        hnext_b[(size_t)m * D_EMB + d] = (bf16)hn;
    }
}

// latent = hT @ Wfc^T + bfc    [64 x 8], fp32 for final precision
__global__ void fc_kernel(const float* __restrict__ h,
                          const float* __restrict__ Wfc,
                          const float* __restrict__ bfc,
                          float* __restrict__ out) {
    int t = threadIdx.x;            // 512 threads
    int b = t >> 3, o = t & 7;
    const float* hr = h + (size_t)b * D_EMB;
    const float* wr = Wfc + (size_t)o * D_EMB;
    float s = bfc[o];
    for (int k = 0; k < D_EMB; ++k) s += hr[k] * wr[k];
    out[b * NB + o] = s;
}

// Spectral norm (3 power iters) + closed-form energy descent:
// pred = -0.5 * (lat @ Wsn^T)
__global__ void sn_pred_kernel(const float* __restrict__ We,
                               const float* __restrict__ u_sn,
                               const float* __restrict__ lat,
                               float* __restrict__ out) {
    __shared__ float Wsn[NB][NB];
    if (threadIdx.x == 0) {
        float u[NB], v[NB];
        float nrm = 0.0f;
        for (int i = 0; i < NB; ++i) { u[i] = u_sn[i]; nrm += u[i] * u[i]; }
        nrm = sqrtf(nrm) + 1e-12f;
        for (int i = 0; i < NB; ++i) u[i] /= nrm;
        for (int it = 0; it < 3; ++it) {
            for (int j = 0; j < NB; ++j) {
                float sv = 0.0f;
                for (int i = 0; i < NB; ++i) sv += We[i * NB + j] * u[i];
                v[j] = sv;
            }
            nrm = 0.0f; for (int j = 0; j < NB; ++j) nrm += v[j] * v[j];
            nrm = sqrtf(nrm) + 1e-12f;
            for (int j = 0; j < NB; ++j) v[j] /= nrm;
            for (int i = 0; i < NB; ++i) {
                float su = 0.0f;
                for (int j = 0; j < NB; ++j) su += We[i * NB + j] * v[j];
                u[i] = su;
            }
            nrm = 0.0f; for (int i = 0; i < NB; ++i) nrm += u[i] * u[i];
            nrm = sqrtf(nrm) + 1e-12f;
            for (int i = 0; i < NB; ++i) u[i] /= nrm;
        }
        float sigma = 0.0f;
        for (int i = 0; i < NB; ++i) {
            float wv = 0.0f;
            for (int j = 0; j < NB; ++j) wv += We[i * NB + j] * v[j];
            sigma += u[i] * wv;
        }
        for (int i = 0; i < NB; ++i)
            for (int j = 0; j < NB; ++j)
                Wsn[i][j] = We[i * NB + j] / sigma;
    }
    __syncthreads();
    int b = threadIdx.x;            // 64 threads
    if (b < BATCH) {
        for (int k = 0; k < NB; ++k) {
            float s = 0.0f;
            for (int l = 0; l < NB; ++l) s += Wsn[k][l] * lat[b * NB + l];
            out[b * NB + k] = -0.5f * s;   // -STEPS*DT * grad, h0 = 0, grad const
        }
    }
}

// ---------------- host ----------------
extern "C" void kernel_launch(void* const* d_in, const int* in_sizes, int n_in,
                              void* d_out, int out_size, void* d_ws, size_t ws_size,
                              hipStream_t stream) {
    const int*   ctx   = (const int*)d_in[0];
    const int*   tgt   = (const int*)d_in[1];
    const float* We    = (const float*)d_in[2];
    const float* u_sn  = (const float*)d_in[3];

    const float* emb   = (const float*)d_in[4];
    const float* Wih   = (const float*)d_in[5];
    const float* Whh   = (const float*)d_in[6];
    const float* bih   = (const float*)d_in[7];
    const float* bhh   = (const float*)d_in[8];
    const float* Wfc   = (const float*)d_in[9];
    const float* bfc   = (const float*)d_in[10];

    const float* temb  = (const float*)d_in[11];
    const float* tWih  = (const float*)d_in[12];
    const float* tWhh  = (const float*)d_in[13];
    const float* tbih  = (const float*)d_in[14];
    const float* tbhh  = (const float*)d_in[15];
    const float* tWfc  = (const float*)d_in[16];
    const float* tbfc  = (const float*)d_in[17];

    float* out = (float*)d_out;

    // workspace carve-up (256B aligned)
    char* ws = (char*)d_ws;
    size_t off = 0;
    auto take = [&](size_t bytes) -> void* {
        void* p = ws + off;
        off += (bytes + 255) & ~(size_t)255;
        return p;
    };
    bf16*  Xb   = (bf16*)take((size_t)ROWS * D_EMB * 2);             // 25 MB
    bf16*  Gi   = (bf16*)take((size_t)ROWS * N3D * 2);               // 75 MB
    bf16*  Pih  = (bf16*)take((size_t)KT_CNT * NT_CNT * 512 * 2);    // 3.5 MB
    bf16*  Phh  = (bf16*)take((size_t)KT_CNT * NT_CNT * 512 * 2);    // 3.5 MB
    float* hf0  = (float*)take((size_t)BATCH * D_EMB * 4);
    float* hf1  = (float*)take((size_t)BATCH * D_EMB * 4);
    bf16*  hb0  = (bf16*)take((size_t)BATCH * D_EMB * 2);
    bf16*  hb1  = (bf16*)take((size_t)BATCH * D_EMB * 2);
    float* latc = (float*)take((size_t)BATCH * NB * 4);

    const int packN = KT_CNT * NT_CNT * 512;
    const int gathN = ROWS * D_EMB;
    const int hN    = BATCH * D_EMB;

    auto run_encoder = [&](const int* tok, const float* e,
                           const float* wih, const float* whh,
                           const float* b_ih, const float* b_hh,
                           const float* wfc, const float* b_fc,
                           float* lat_out) {
        pack_w_kernel<<<(packN + 255) / 256, 256, 0, stream>>>(wih, Pih);
        pack_w_kernel<<<(packN + 255) / 256, 256, 0, stream>>>(whh, Phh);
        gather_kernel<<<(gathN + 255) / 256, 256, 0, stream>>>(tok, e, Xb);
        gi_gemm_kernel<<<dim3(ROWS / 64, NT_CNT), 128, 0, stream>>>(Xb, Pih, b_ih, Gi);
        zero_h_kernel<<<(hN + 255) / 256, 256, 0, stream>>>(hf0, hb0);
        for (int s = 0; s < SEQ; ++s) {
            const float* hpf = (s & 1) ? hf1 : hf0;
            const bf16*  hpb = (s & 1) ? hb1 : hb0;
            float*       hnf = (s & 1) ? hf0 : hf1;
            bf16*        hnb = (s & 1) ? hb0 : hb1;
            gru_step_kernel<<<DT_CNT, 128, 0, stream>>>(Gi, s, hpf, hpb, Phh, b_hh, hnf, hnb);
        }
        // SEQ=256 even -> final hidden state lands back in buffer 0
        fc_kernel<<<1, 512, 0, stream>>>(hf0, wfc, b_fc, lat_out);
    };

    // context encoder -> pred_latent (closed-form energy descent) -> out[0..511]
    run_encoder(ctx, emb, Wih, Whh, bih, bhh, Wfc, bfc, latc);
    sn_pred_kernel<<<1, 64, 0, stream>>>(We, u_sn, latc, out);

    // target encoder -> target_latent -> out[512..1023]
    run_encoder(tgt, temb, tWih, tWhh, tbih, tbhh, tWfc, tbfc, out + BATCH * NB);
}
